// RLPatchPolicy_31044023616503
// MI455X (gfx1250) — compile-verified, matrix operand order checked by reading, not run on hardware
//
#include <hip/hip_runtime.h>
#include <hip/hip_bf16.h>
#include <math.h>

// ---------------------------------------------------------------------------
// Types for CDNA5 WMMA (wave32): v_wmma_f32_16x16x32_f16
// ---------------------------------------------------------------------------
typedef __attribute__((ext_vector_type(16))) _Float16 v16h;
typedef __attribute__((ext_vector_type(8)))  _Float16 v8h;   // 16 bytes
typedef __attribute__((ext_vector_type(8)))  float    v8f;

union HF16 { v16h v; v8h h[2]; };
union U8H  { v8h v; _Float16 e[8]; };

#define EPS_BN 1e-5f

// ---------------------------------------------------------------------------
// Small utility kernels
// ---------------------------------------------------------------------------
__global__ void k_cvt_f32_to_f16(const float* __restrict__ in,
                                 _Float16* __restrict__ out, int n) {
  for (int i = blockIdx.x * blockDim.x + threadIdx.x; i < n;
       i += gridDim.x * blockDim.x)
    out[i] = (_Float16)in[i];
}

// repack conv weights (Cout, Cin, 3, 3) fp32 -> f16 [tap][cout][cin]
__global__ void k_pack_w(const float* __restrict__ w,
                         _Float16* __restrict__ wp, int Cout, int Cin) {
  int total = 9 * Cout * Cin;
  for (int o = blockIdx.x * blockDim.x + threadIdx.x; o < total;
       o += gridDim.x * blockDim.x) {
    int tap = o / (Cout * Cin);
    int rem = o - tap * (Cout * Cin);
    int m = rem / Cin;
    int c = rem - m * Cin;
    wp[o] = (_Float16)w[(size_t)(m * Cin + c) * 9 + tap];
  }
}

__global__ void k_bn_prep(const float* __restrict__ g, const float* __restrict__ be,
                          const float* __restrict__ mn, const float* __restrict__ vr,
                          const float* __restrict__ convb,
                          float* __restrict__ sc, float* __restrict__ sh, int C) {
  int i = blockIdx.x * blockDim.x + threadIdx.x;
  if (i < C) {
    float s = g[i] * rsqrtf(vr[i] + EPS_BN);
    sc[i] = s;
    sh[i] = (convb[i] - mn[i]) * s + be[i];
  }
}

__global__ void k_zero_f32(float* __restrict__ p, int n) {
  for (int i = blockIdx.x * blockDim.x + threadIdx.x; i < n;
       i += gridDim.x * blockDim.x)
    p[i] = 0.0f;
}

__global__ void k_clin(const float* __restrict__ clin, const float* __restrict__ w,
                       const float* __restrict__ b, float* __restrict__ pooled) {
  int i = blockIdx.x * blockDim.x + threadIdx.x;
  if (i < 16 * 128) {
    int bb = i >> 7, j = i & 127;
    float s = b[j];
#pragma unroll
    for (int t = 0; t < 4; ++t) s += clin[bb * 4 + t] * w[j * 4 + t];
    pooled[bb * 8320 + 8192 + j] = s;
  }
}

// ---------------------------------------------------------------------------
// Implicit-GEMM 3x3 SAME conv as 9 shifted GEMMs, f16 WMMA, f32 accumulate.
// Block tile 128(M) x 256(N), 8 waves, each wave 64x64 = 4x4 16x16 tiles
// (16 WMMAs per K-slab vs 16 ds_load_b128 -> 1:1 LDS feed ratio).
// Double-buffered LDS, single barrier per slab:
//   per slab:  ds_store(regs) ; barrier ; global-load next slab -> regs ;
//              16 WMMAs on current buffer   (loads fly under the WMMAs)
// ---------------------------------------------------------------------------
__global__ __launch_bounds__(256) void k_conv3x3_wmma(
    const _Float16* __restrict__ xin,   // [B*Cin*4096] NCHW f16
    const _Float16* __restrict__ wp,    // [9][Cout][Cin] f16
    const float* __restrict__ bnscale, const float* __restrict__ bnshift,
    _Float16* __restrict__ xout,        // mode 0
    float* __restrict__ pooled,         // mode 1
    int Cin, int mode) {
  const int Cout = 512;
  // ping-pong buffers, pitch 40 halves (aligned b128 fragment reads)
  __shared__ __align__(16) _Float16 As[2][128 * 40];   // 2 x 10 KB
  __shared__ __align__(16) _Float16 Bs[2][256 * 40];   // 2 x 20 KB

  const int tid  = threadIdx.x;
  const int lane = tid & 31;
  const int wv   = tid >> 5;
  const int wm   = wv >> 2;       // 0..1 : M offset wm*64
  const int wn   = wv & 3;        // 0..3 : N offset wn*64
  const int n0   = blockIdx.x * 256;
  const int m0   = blockIdx.y * 128;

  v8f acc[4][4] = {};

  // A-slab loader: 128 m-rows x 32 k, 16 halves per thread
  const int a_m = tid >> 1;
  const int a_k = (tid & 1) * 16;
  // B-slab loader: k = tid>>3 (0..31); 32-column row segment per thread
  const int b_k   = tid >> 3;
  const int seg   = tid & 7;
  const int r_loc = seg >> 1;          // row 0..3 within 4-row N tile
  const int wl0   = (seg & 1) << 5;    // 0 or 32
  const int nl0   = seg << 5;          // first dest column in tile (seg*32)
  const int b_img = n0 >> 12;
  const int row0  = (n0 & 4095) >> 6;

  const int lm = lane & 15;
  const int kb = (lane >> 4) * 8;

  const int npt   = Cin >> 5;          // K-slabs per tap
  const int nslab = 9 * npt;

  // ---- staging registers (one slab in flight) ----
  v8h a_lo, a_hi;                      // weights
  U8H bm[4];                           // activations: 32 aligned halves
  _Float16 bl, br;                     // edge halves for dc = -1 / +1
  int dc_cur = 0;

  auto gload = [&](int tap, int c0) {
    const int dr = tap / 3 - 1;
    dc_cur = tap % 3 - 1;
    const _Float16* ga = wp + ((size_t)tap * Cout + (m0 + a_m)) * Cin + c0 + a_k;
    a_lo = *(const v8h*)ga;
    a_hi = *(const v8h*)(ga + 8);
    if (c0 + 32 < Cin)  // gfx1250 global_prefetch_b8 on the following slab
      __builtin_prefetch(ga + 32, 0, 1);
    const int srow = row0 + r_loc + dr;
    v8h z = {};
#pragma unroll
    for (int q = 0; q < 4; ++q) bm[q].v = z;
    bl = (_Float16)0.0f; br = (_Float16)0.0f;
    if ((unsigned)srow < 64u) {
      const _Float16* src =
          xin + ((size_t)(b_img * Cin + c0 + b_k) << 12) + (srow << 6);
#pragma unroll
      for (int q = 0; q < 4; ++q)
        bm[q].v = *(const v8h*)(src + wl0 + q * 8);
      if (wl0 > 0)       bl = src[wl0 - 1];
      if (wl0 + 32 < 64) br = src[wl0 + 32];
    }
  };

  auto sstore = [&](int buf) {
    *(v8h*)&As[buf][a_m * 40 + a_k]     = a_lo;
    *(v8h*)&As[buf][a_m * 40 + a_k + 8] = a_hi;
    _Float16* bp = &Bs[buf][nl0 * 40 + b_k];
    if (dc_cur == 0) {
#pragma unroll
      for (int j = 0; j < 32; ++j) bp[j * 40] = bm[j >> 3].e[j & 7];
    } else if (dc_cur == 1) {       // dst j <- src wl0 + j + 1
#pragma unroll
      for (int j = 0; j < 31; ++j) bp[j * 40] = bm[(j + 1) >> 3].e[(j + 1) & 7];
      bp[31 * 40] = br;
    } else {                        // dst j <- src wl0 + j - 1
      bp[0] = bl;
#pragma unroll
      for (int j = 1; j < 32; ++j) bp[j * 40] = bm[(j - 1) >> 3].e[(j - 1) & 7];
    }
  };

  auto compute = [&](int buf) {
    HF16 bf[4];
#pragma unroll
    for (int ns = 0; ns < 4; ++ns) {
      int nrow = wn * 64 + ns * 16 + lm;
      bf[ns].h[0] = *(const v8h*)&Bs[buf][nrow * 40 + kb];
      bf[ns].h[1] = *(const v8h*)&Bs[buf][nrow * 40 + kb + 16];
    }
#pragma unroll
    for (int ms = 0; ms < 4; ++ms) {
      HF16 af;
      int mrow = wm * 64 + ms * 16 + lm;
      af.h[0] = *(const v8h*)&As[buf][mrow * 40 + kb];
      af.h[1] = *(const v8h*)&As[buf][mrow * 40 + kb + 16];
#pragma unroll
      for (int ns = 0; ns < 4; ++ns) {
        acc[ms][ns] = __builtin_amdgcn_wmma_f32_16x16x32_f16(
            false, af.v, false, bf[ns].v, (short)0, acc[ms][ns],
            false, false);
      }
    }
  };

  // ---- software pipeline, one barrier per slab ----
  int tap_n = 0, c0_n = 0;
  gload(tap_n, c0_n);                       // slab 0 -> regs
  for (int s = 0; s < nslab; ++s) {
    const int buf = s & 1;
    sstore(buf);                            // regs (slab s) -> LDS
    __syncthreads();                        // staging visible; prev compute done
    c0_n += 32;
    if (c0_n == Cin) { c0_n = 0; ++tap_n; }
    if (s + 1 < nslab) gload(tap_n, c0_n);  // issue slab s+1 loads
    compute(buf);                           // WMMAs overlap the loads
  }

  // ---- epilogue: C/D layout: VGPR j -> row = j + 8*(lane>>4), col = lane&15
  const int mhi = (lane >> 4) * 8;
  if (mode == 0) {
#pragma unroll
    for (int ms = 0; ms < 4; ++ms)
#pragma unroll
      for (int ns = 0; ns < 4; ++ns)
#pragma unroll
        for (int j = 0; j < 8; ++j) {
          int m = m0 + wm * 64 + ms * 16 + mhi + j;
          int n = n0 + wn * 64 + ns * 16 + lm;
          float v = acc[ms][ns][j] * bnscale[m] + bnshift[m];
          v = fmaxf(v, 0.0f);
          int bi = n >> 12, pos = n & 4095;
          xout[(size_t)(bi * Cout + m) * 4096 + pos] = (_Float16)v;
        }
  } else {
#pragma unroll
    for (int ms = 0; ms < 4; ++ms)
#pragma unroll
      for (int ns = 0; ns < 4; ++ns)
#pragma unroll
        for (int j = 0; j < 8; ++j) {
          int m = m0 + wm * 64 + ms * 16 + mhi + j;
          int n = n0 + wn * 64 + ns * 16 + lm;
          float v = acc[ms][ns][j] * bnscale[m] + bnshift[m];
          v = fmaxf(v, 0.0f);
          for (int mk = 8; mk >= 1; mk >>= 1) v += __shfl_xor(v, mk, 32);
          if (lm == 0) {
            int bi = n >> 12, pos = n & 4095;
            int ph = (pos >> 6) >> 4, pw = (pos & 63) >> 4;
            atomicAdd(&pooled[bi * 8320 + m * 16 + ph * 4 + pw],
                      v * (1.0f / 256.0f));
          }
        }
  }
}

// ---------------------------------------------------------------------------
// Wave-per-output linear with float4 loads: out[b][o] = act(dot + bias)
// act: 0 = relu, 1 = none, 2 = sigmoid.  Kd must be a multiple of 4.
// ---------------------------------------------------------------------------
__global__ void k_lin_act(const float* __restrict__ in, int in_pitch,
                          const float* __restrict__ W, const float* __restrict__ bias,
                          float* __restrict__ out, int out_pitch,
                          int B, int O, int Kd, int act) {
  int gw = (blockIdx.x * blockDim.x + threadIdx.x) >> 5;
  int lane = threadIdx.x & 31;
  if (gw >= B * O) return;
  int b = gw / O, o = gw - b * O;
  const float4* x4 = (const float4*)(in + (size_t)b * in_pitch);
  const float4* w4 = (const float4*)(W + (size_t)o * Kd);
  int k4 = Kd >> 2;
  float s = 0.0f;
  for (int k = lane; k < k4; k += 32) {
    float4 a = x4[k], w = w4[k];
    s += a.x * w.x + a.y * w.y + a.z * w.z + a.w * w.w;
  }
  for (int m = 16; m >= 1; m >>= 1) s += __shfl_xor(s, m, 32);
  if (lane == 0) {
    s += bias[o];
    if (act == 0) s = fmaxf(s, 0.0f);
    else if (act == 2) s = 1.0f / (1.0f + expf(-s));
    out[(size_t)b * out_pitch + o] = s;
  }
}

// ---------------------------------------------------------------------------
// Reparameterize + log-probs + point sampling.
// out = [coords(480) | log_probs(48) | patches(192)]
// ---------------------------------------------------------------------------
__global__ void k_finalize(const float* __restrict__ pm,   // [16][12] sigmoid
                           const float* __restrict__ lsd,  // [16][12] raw
                           const float* __restrict__ noise,// (16,3,4)
                           const float* __restrict__ u,    // (16,3,5,2)
                           float* __restrict__ out) {
  int tid = blockIdx.x * blockDim.x + threadIdx.x;
  if (tid >= 48) return;
  int b = tid / 3, k = tid - b * 3;
  float pat[4], logp = 0.0f;
#pragma unroll
  for (int j = 0; j < 4; ++j) {
    int idx = b * 12 + k * 4 + j;
    float m = pm[idx];
    float ls = fminf(fmaxf(lsd[idx], -5.0f), 0.0f);
    float sd = expf(ls);
    float p = m + noise[idx] * sd;
    float z = (p - m) / sd;
    logp += z * z + 2.0f * ls;
    pat[j] = fminf(fmaxf(p, 0.0f), 1.0f);
  }
  out[480 + b * 3 + k] = -0.5f * logp;
  float sx = pat[2] * 0.2f + 0.1f;
  float sy = pat[3] * 0.2f + 0.1f;
  int po = 528 + (b * 3 + k) * 4;
  out[po + 0] = pat[0];
  out[po + 1] = pat[1];
  out[po + 2] = sx;
  out[po + 3] = sy;
#pragma unroll
  for (int pi = 0; pi < 5; ++pi) {
    int ui = ((b * 3 + k) * 5 + pi) * 2;
    float dx = (u[ui + 0] - 0.5f) * sx;
    float dy = (u[ui + 1] - 0.5f) * sy;
    float px = fminf(fmaxf((pat[0] + dx) * 256.0f, 0.0f), 255.0f);
    float py = fminf(fmaxf((pat[1] + dy) * 256.0f, 0.0f), 255.0f);
    int co = (b * 15 + k * 5 + pi) * 2;
    out[co + 0] = px;
    out[co + 1] = py;
  }
}

// ---------------------------------------------------------------------------
// Host launcher
// ---------------------------------------------------------------------------
extern "C" void kernel_launch(void* const* d_in, const int* in_sizes, int n_in,
                              void* d_out, int out_size, void* d_ws, size_t ws_size,
                              hipStream_t stream) {
  (void)in_sizes; (void)n_in; (void)out_size; (void)ws_size;
  const float* img     = (const float*)d_in[0];
  const float* clin    = (const float*)d_in[1];
  const float* noise   = (const float*)d_in[2];
  const float* u       = (const float*)d_in[3];
  const float* conv1_w = (const float*)d_in[4];
  const float* conv1_b = (const float*)d_in[5];
  const float* bn1_g   = (const float*)d_in[6];
  const float* bn1_b   = (const float*)d_in[7];
  const float* bn1_m   = (const float*)d_in[8];
  const float* bn1_v   = (const float*)d_in[9];
  const float* conv2_w = (const float*)d_in[10];
  const float* conv2_b = (const float*)d_in[11];
  const float* bn2_g   = (const float*)d_in[12];
  const float* bn2_b   = (const float*)d_in[13];
  const float* bn2_m   = (const float*)d_in[14];
  const float* bn2_v   = (const float*)d_in[15];
  const float* clin_w  = (const float*)d_in[16];
  const float* clin_b  = (const float*)d_in[17];
  const float* ph1_w   = (const float*)d_in[18];
  const float* ph1_b   = (const float*)d_in[19];
  const float* ph2_w   = (const float*)d_in[20];
  const float* ph2_b   = (const float*)d_in[21];
  const float* ph3_w   = (const float*)d_in[22];
  const float* ph3_b   = (const float*)d_in[23];
  const float* ls1_w   = (const float*)d_in[24];
  const float* ls1_b   = (const float*)d_in[25];
  const float* ls2_w   = (const float*)d_in[26];
  const float* ls2_b   = (const float*)d_in[27];
  float* out = (float*)d_out;

  char* ws = (char*)d_ws;
  size_t off = 0;
  auto carve = [&](size_t bytes) -> char* {
    char* p = ws + off;
    off = (off + bytes + 255) & ~(size_t)255;
    return p;
  };
  _Float16* img_h = (_Float16*)carve((size_t)16 * 256 * 4096 * 2);  // 32 MB
  _Float16* x1h   = (_Float16*)carve((size_t)16 * 512 * 4096 * 2);  // 64 MB
  _Float16* w1p   = (_Float16*)carve((size_t)9 * 512 * 256 * 2);
  _Float16* w2p   = (_Float16*)carve((size_t)9 * 512 * 512 * 2);
  float* bnsc1    = (float*)carve(512 * 4);
  float* bnsh1    = (float*)carve(512 * 4);
  float* bnsc2    = (float*)carve(512 * 4);
  float* bnsh2    = (float*)carve(512 * 4);
  float* pooled   = (float*)carve((size_t)16 * 8320 * 4);
  float* h1       = (float*)carve((size_t)16 * 512 * 4);
  float* h2       = (float*)carve((size_t)16 * 256 * 4);
  float* g1       = (float*)carve((size_t)16 * 256 * 4);
  float* pm       = (float*)carve((size_t)16 * 12 * 4);
  float* lsd      = (float*)carve((size_t)16 * 12 * 4);

  k_cvt_f32_to_f16<<<8192, 256, 0, stream>>>(img, img_h, 16 * 256 * 4096);
  k_pack_w<<<2048, 256, 0, stream>>>(conv1_w, w1p, 512, 256);
  k_pack_w<<<4096, 256, 0, stream>>>(conv2_w, w2p, 512, 512);
  k_bn_prep<<<2, 256, 0, stream>>>(bn1_g, bn1_b, bn1_m, bn1_v, conv1_b, bnsc1, bnsh1, 512);
  k_bn_prep<<<2, 256, 0, stream>>>(bn2_g, bn2_b, bn2_m, bn2_v, conv2_b, bnsc2, bnsh2, 512);
  k_zero_f32<<<520, 256, 0, stream>>>(pooled, 16 * 8320);
  k_clin<<<8, 256, 0, stream>>>(clin, clin_w, clin_b, pooled);

  dim3 cgrid(256, 4);   // N tiles (65536/256) x M tiles (512/128)
  k_conv3x3_wmma<<<cgrid, 256, 0, stream>>>(img_h, w1p, bnsc1, bnsh1,
                                            x1h, nullptr, 256, 0);
  k_conv3x3_wmma<<<cgrid, 256, 0, stream>>>(x1h, w2p, bnsc2, bnsh2,
                                            nullptr, pooled, 512, 1);

  k_lin_act<<<1024, 256, 0, stream>>>(pooled, 8320, ph1_w, ph1_b, h1, 512,
                                      16, 512, 8320, 0);
  k_lin_act<<<512, 256, 0, stream>>>(h1, 512, ph2_w, ph2_b, h2, 256,
                                     16, 256, 512, 0);
  k_lin_act<<<24, 256, 0, stream>>>(h2, 256, ph3_w, ph3_b, pm, 12,
                                    16, 12, 256, 2);
  k_lin_act<<<512, 256, 0, stream>>>(pooled, 8320, ls1_w, ls1_b, g1, 256,
                                     16, 256, 8320, 0);
  k_lin_act<<<24, 256, 0, stream>>>(g1, 256, ls2_w, ls2_b, lsd, 12,
                                    16, 12, 256, 1);

  k_finalize<<<1, 64, 0, stream>>>(pm, lsd, noise, u, out);
}
